// MPGCN_87033217286372
// MI455X (gfx1250) — matrix operand order repaired and checked by reading, not compiled, not run on hardware
//
#include <hip/hip_runtime.h>
#include <hip/hip_bf16.h>

// ---------------- CDNA5 WMMA types ----------------
typedef __attribute__((ext_vector_type(16))) __bf16 v16bf;
typedef __attribute__((ext_vector_type(8)))  float  v8f;
typedef __attribute__((ext_vector_type(2)))  __bf16 bf16x2;
typedef __attribute__((ext_vector_type(4)))  unsigned int u32x4;
typedef __attribute__((ext_vector_type(8)))  int i32x8;
typedef __attribute__((ext_vector_type(4)))  int i32x4;

// ---------------- problem constants ----------------
#define NN   64
#define HH   64
#define TT   12
#define BB   8
#define KK   3
#define MM   2
#define RTOT (BB*NN*NN)        // 32768 sequences
#define XPB  (BB*NN*NN*HH)     // 2097152 f32 elems per-branch activation

__device__ __forceinline__ v8f wmma_bf16(v16bf a, v16bf b, v8f c) {
  // D(16x16,f32) = A(16x32,bf16) x B(32x16,bf16) + C
  return __builtin_amdgcn_wmma_f32_16x16x32_bf16(false, a, false, b,
                                                 (short)0, c, false, false);
}

// ---- fast activations on native gfx1250 transcendentals ----
#define LOG2E 1.4426950408889634f
__device__ __forceinline__ float fast_sigmoid(float x) {
  float e = __builtin_amdgcn_exp2f(-LOG2E * x);          // exp(-x)
  return __builtin_amdgcn_rcpf(1.0f + e);
}
__device__ __forceinline__ float fast_tanh(float x) {
#if __has_builtin(__builtin_amdgcn_tanhf)
  return __builtin_amdgcn_tanhf(x);                      // CDNA5 V_TANH_F32
#elif __has_builtin(__builtin_amdgcn_tanh_f32)
  return __builtin_amdgcn_tanh_f32(x);
#else
  float e = __builtin_amdgcn_exp2f(-2.0f * LOG2E * x);   // exp(-2x)
  return 2.0f * __builtin_amdgcn_rcpf(1.0f + e) - 1.0f;
#endif
}

// Gather a 16x32 A-tile. `base` points at row m=0 of a [16][64] bf16 region
// (row stride 64); koff selects K window {0,32}. Layout per ISA 7.12.2:
// lane L: m=L&15; VGPR v holds K = 16*(v>>2) + 8*(L>>4) + 2*(v&3) (+1).
__device__ __forceinline__ v16bf gather_A(const __bf16* base, int koff) {
  const int lane = threadIdx.x & 31;
  const int m = lane & 15, hi = lane >> 4;
  v16bf a;
#pragma unroll
  for (int v = 0; v < 8; ++v) {
    int k = koff + 16*(v>>2) + 8*hi + 2*(v&3);
    bf16x2 p = *(const bf16x2*)(base + m*64 + k);
    a[2*v] = p.x; a[2*v+1] = p.y;
  }
  return a;
}

// Gather a 32x16 B-tile from an [n][k] bf16 layout (row stride 64).
// lane L: n=n0+(L&15); VGPR v holds K = k0 + 16*(v>>2) + 8*(L>>4) + 2*(v&3) (+1).
__device__ __forceinline__ v16bf gather_B(const __bf16* base, int n0, int k0) {
  const int lane = threadIdx.x & 31;
  const int n = n0 + (lane & 15), hi = lane >> 4;
  v16bf b;
#pragma unroll
  for (int v = 0; v < 8; ++v) {
    int k = k0 + 16*(v>>2) + 8*hi + 2*(v&3);
    bf16x2 p = *(const bf16x2*)(base + n*64 + k);
    b[2*v] = p.x; b[2*v+1] = p.y;
  }
  return b;
}

// ---- Tensor Data Mover: 1-D 8KB bf16 tile, global -> LDS ----
// D# per ISA 8.3/8.4: group0 = {count=1, lds_addr, global_addr, type=2};
// group1 = {data_size=2B, tensor_dim0=tile_dim0=4096, dim1=1, stride0=4096}.
#if __has_builtin(__builtin_amdgcn_tensor_load_to_lds)
#define HAVE_TDM 1
__device__ __forceinline__ void tdm_load_8kb(const __bf16* gsrc, __bf16* lds_dst) {
  unsigned long long ga = (unsigned long long)(size_t)gsrc;
  unsigned lds = (unsigned)(size_t)lds_dst;     // low 32 bits = LDS byte offset
  u32x4 g0;
  g0[0] = 1u;                                               // count=1 (user D#)
  g0[1] = lds;                                              // lds_addr
  g0[2] = (unsigned)(ga & 0xffffffffull);                   // global_addr[31:0]
  g0[3] = (unsigned)((ga >> 32) & 0x1ffffffull) | (2u << 30); // [56:32] | type=2
  i32x8 g1;
  g1[0] = (int)(1u << 16);                 // data_size=1 (2 bytes), mask=0
  g1[1] = (int)(4096u << 16);              // tensor_dim0[15:0] in bits 63:48
  g1[2] = (int)(1u << 16);                 // tensor_dim0 hi=0; tensor_dim1=1
  g1[3] = (int)(4096u << 16);              // tile_dim0=4096 (bits 127:112)
  g1[4] = 1;                               // tile_dim1=1, tile_dim2=0
  g1[5] = 4096;                            // tensor_dim0_stride lo32
  g1[6] = 0;
  g1[7] = 0;
  i32x4 z4 = {0, 0, 0, 0};
#if defined(__clang_major__) && (__clang_major__ >= 23)
  i32x8 z8 = {0, 0, 0, 0, 0, 0, 0, 0};
  __builtin_amdgcn_tensor_load_to_lds(g0, g1, z4, z4, z8, 0);
#else
  __builtin_amdgcn_tensor_load_to_lds(g0, g1, z4, z4, 0);
#endif
}
#endif

// =====================================================================
// Kernel 1: LSTM over T steps. One wave per 16-row tile; block = 8 waves.
// gates(16x256) = h(16x64) @ Whh^T + x*Wih + bias, processed as 4 column
// groups so (i,f,g,o) tiles for the same 16 h-columns are live together.
// c stays in D-layout registers; h round-trips wave-private LDS as bf16.
// =====================================================================
__global__ __launch_bounds__(256) void lstm_kernel(
    const float* __restrict__ x_seq, const float* __restrict__ Wih,
    const float* __restrict__ Whh,  const float* __restrict__ bih,
    const float* __restrict__ bhh,  float* __restrict__ Hbuf)
{
  __shared__ __bf16 WhhL[256*64];        // [n][k] == Whh row-major, 32KB
  __shared__ float  WihL[256];
  __shared__ float  biasL[256];
  __shared__ __bf16 hL[8][16*64];        // per-wave h, bf16, 16KB
  __shared__ float  xv[8][16];           // per-wave x_t

  const int mb  = blockIdx.y;
  const int tid = threadIdx.x;
  const int w = tid >> 5, lane = tid & 31;
  const int hi = lane >> 4, lo = lane & 15;

  const float* WhhG = Whh + (size_t)mb*256*64;
  for (int i = tid; i < 256*64; i += 256) WhhL[i] = (__bf16)WhhG[i];
  WihL[tid]  = Wih[mb*256 + tid];
  biasL[tid] = bih[mb*256 + tid] + bhh[mb*256 + tid];
  for (int i = lane; i < 16*64; i += 32) hL[w][i] = (__bf16)0.0f;
  __syncthreads();

  const int rowbase = (blockIdx.x*8 + w) * 16;   // first of 16 sequences

  v8f c[4];
#pragma unroll
  for (int t4 = 0; t4 < 4; ++t4)
#pragma unroll
    for (int v = 0; v < 8; ++v) c[t4][v] = 0.0f;

  for (int t = 0; t < TT; ++t) {
    if (lane < 16) {                     // x_t for my 16 rows (coalesced)
      int r = rowbase + lane;
      int b = r >> 12, od = r & 4095;
      xv[w][lane] = x_seq[(size_t)b*(TT*4096) + (size_t)t*4096 + od];
    }
    v16bf a0 = gather_A(&hL[w][0], 0);   // h cols 0..31
    v16bf a1 = gather_A(&hL[w][0], 32);  // h cols 32..63

#pragma unroll
    for (int t4 = 0; t4 < 4; ++t4) {     // 16 h-columns per group
      v8f acc[4];                        // i,f,g,o tiles
#pragma unroll
      for (int g = 0; g < 4; ++g) {
        int nt = t4 + 4*g;               // gate n-tile
        v16bf b0 = gather_B(WhhL, nt*16, 0);
        v16bf b1 = gather_B(WhhL, nt*16, 32);
        v8f z;
#pragma unroll
        for (int v = 0; v < 8; ++v) z[v] = 0.0f;
        z = wmma_bf16(a0, b0, z);
        z = wmma_bf16(a1, b1, z);
        acc[g] = z;
      }
#pragma unroll
      for (int v = 0; v < 8; ++v) {
        int m   = v + 8*hi;              // row within tile (D layout)
        int col = t4*16 + lo;            // h column
        float xr = xv[w][m];
        float gi = acc[0][v] + xr*WihL[col]       + biasL[col];
        float gf = acc[1][v] + xr*WihL[64  + col] + biasL[64  + col];
        float gg = acc[2][v] + xr*WihL[128 + col] + biasL[128 + col];
        float go = acc[3][v] + xr*WihL[192 + col] + biasL[192 + col];
        float cn = fast_sigmoid(gf)*c[t4][v] + fast_sigmoid(gi)*fast_tanh(gg);
        c[t4][v] = cn;
        float h  = fast_sigmoid(go)*fast_tanh(cn);
        hL[w][m*64 + col] = (__bf16)h;   // bf16 recurrence state
        if (t == TT-1)
          Hbuf[(size_t)mb*XPB + (size_t)(rowbase + m)*64 + col] = h;
      }
    }
  }
}

// =====================================================================
// Kernel 2: T1[o,b](m, c, l) = G[o]^T(m,n) @ X[b](n, c*64+l), written to
// scratch transposed as bf16 [m][l][c] so the fused kernel's B-operand
// ([l][c] pairs over c) is a straight copy. Block = 4 waves, one 64-col
// chunk (==> c is constant per block), all 64 m rows.
// =====================================================================
__global__ __launch_bounds__(128) void gcn_t1_kernel(
    const float* __restrict__ X, const float* __restrict__ G,
    __bf16* __restrict__ T1g)
{
  __shared__ __bf16 GoT[64*64];   // [m][n] = G[o]^T
  __shared__ __bf16 Xt [64*64];   // [col][n]
  const int chunk = blockIdx.x;               // c index (0..63)
  const int o = blockIdx.y / BB, b = blockIdx.y % BB;
  const int mb  = blockIdx.z;
  const int tid = threadIdx.x;
  const int w = tid >> 5, lane = tid & 31;
  const int hi = lane >> 4, lo = lane & 15;

  const float* Go = G + ((size_t)mb*KK + o)*4096;
  for (int i = tid; i < 4096; i += 128) {     // coalesced read, LDS transpose
    int n = i >> 6, mrow = i & 63;
    GoT[mrow*64 + n] = (__bf16)Go[i];
  }
  const float* Xb = X + (size_t)mb*XPB + (size_t)b*(NN*NN*HH);
  for (int i = tid; i < 4096; i += 128) {
    int n = i >> 6, col = i & 63;
    Xt[col*64 + n] = (__bf16)Xb[(size_t)n*4096 + chunk*64 + col];
  }
  __syncthreads();

  __bf16* out = T1g + ((size_t)(mb*KK*BB + o*BB + b))*(64*64*64); // [m][l][c]
  v16bf B0 = gather_B(Xt, w*16, 0);
  v16bf B1 = gather_B(Xt, w*16, 32);
#pragma unroll
  for (int mt = 0; mt < 4; ++mt) {
    v16bf A0 = gather_A(GoT + mt*16*64, 0);
    v16bf A1 = gather_A(GoT + mt*16*64, 32);
    v8f acc;
#pragma unroll
    for (int v = 0; v < 8; ++v) acc[v] = 0.0f;
    acc = wmma_bf16(A0, B0, acc);
    acc = wmma_bf16(A1, B1, acc);
#pragma unroll
    for (int v = 0; v < 8; ++v) {
      int m = mt*16 + v + 8*hi;
      int l = w*16 + lo;                       // col within chunk == l
      out[(size_t)m*4096 + l*64 + chunk] = (__bf16)acc[v];
    }
  }
}

// =====================================================================
// Kernel 3 (fused): for each (b, node m) accumulate over the 9 (o,d):
//   tmp(e,l) = G[d]^T @ T1[o,b,m]     (through LDS, bf16)
//   out(e,h) += tmp @ W[(o*K+d)*64 + l][h]
// T1 tile (8KB bf16, contiguous) is staged by the Tensor Data Mover from
// wave 0 (EXEC-independent DMA, tracked with TENSORcnt); the other waves
// stage the f32->bf16 converted G/W operands meanwhile. Block = 4 waves;
// wave w owns e-rows 16w..16w+15 so tmp A-gathers are wave-private.
// =====================================================================
__global__ __launch_bounds__(128) void gcn_out_kernel(
    const __bf16* __restrict__ T1g, const float* __restrict__ G,
    const float* __restrict__ W,    const float* __restrict__ bias,
    float* __restrict__ Y)
{
  __shared__ __bf16 GdT[64*64];  // [e][c] = G[d]^T
  __shared__ __bf16 T1c[64*64];  // [l][c]
  __shared__ __bf16 Wt [64*64];  // [h][l]
  __shared__ __bf16 tmp[64*64];  // [e][l]
  const int mrow = blockIdx.x;               // node index m
  const int b    = blockIdx.y;
  const int mb   = blockIdx.z;
  const int tid = threadIdx.x;
  const int w = tid >> 5, lane = tid & 31;
  const int hi = lane >> 4, lo = lane & 15;

  v8f acc[4];
#pragma unroll
  for (int ht = 0; ht < 4; ++ht)
#pragma unroll
    for (int v = 0; v < 8; ++v) acc[ht][v] = 0.0f;

  for (int o = 0; o < KK; ++o) {
    const __bf16* T1m = T1g + ((size_t)(mb*KK*BB + o*BB + b))*(64*64*64)
                            + (size_t)mrow*4096;          // [l][c] bf16, 8KB
#ifdef HAVE_TDM
    if (w == 0) {                          // one wave drives the TDM
      tdm_load_8kb(T1m, T1c);
      __builtin_amdgcn_s_wait_tensorcnt(0);
    }
#else
    for (int i = tid; i < 2048; i += 128)  // fallback: dword copy
      ((unsigned int*)T1c)[i] = ((const unsigned int*)T1m)[i];
#endif

    for (int d = 0; d < KK; ++d) {
      const float* Gd = G + ((size_t)mb*KK + d)*4096;
      for (int i = tid; i < 4096; i += 128) {
        int cc = i >> 6, e = i & 63;
        GdT[e*64 + cc] = (__bf16)Gd[i];
      }
      const float* Wod = W + (size_t)((o*KK + d)*64)*64;   // rows l, cols h
      for (int i = tid; i < 4096; i += 128) {
        int l = i >> 6, h = i & 63;
        Wt[h*64 + l] = (__bf16)Wod[i];
      }
      __syncthreads();                     // publishes T1c + GdT + Wt

      // tmp stripe: e rows 16w..16w+15, all l
      v16bf Ag0 = gather_A(GdT + w*16*64, 0);
      v16bf Ag1 = gather_A(GdT + w*16*64, 32);
#pragma unroll
      for (int lt = 0; lt < 4; ++lt) {
        v16bf B0 = gather_B(T1c, lt*16, 0);
        v16bf B1 = gather_B(T1c, lt*16, 32);
        v8f z;
#pragma unroll
        for (int v = 0; v < 8; ++v) z[v] = 0.0f;
        z = wmma_bf16(Ag0, B0, z);
        z = wmma_bf16(Ag1, B1, z);
#pragma unroll
        for (int v = 0; v < 8; ++v)
          tmp[(w*16 + v + 8*hi)*64 + lt*16 + lo] = (__bf16)z[v];
      }
      // out(e,h) += tmp(e,l) @ W[od](l,h)   (wave-private tmp rows)
      v16bf At0 = gather_A(tmp + w*16*64, 0);
      v16bf At1 = gather_A(tmp + w*16*64, 32);
#pragma unroll
      for (int ht = 0; ht < 4; ++ht) {
        v16bf B0 = gather_B(Wt, ht*16, 0);
        v16bf B1 = gather_B(Wt, ht*16, 32);
        acc[ht] = wmma_bf16(At0, B0, acc[ht]);
        acc[ht] = wmma_bf16(At1, B1, acc[ht]);
      }
      __syncthreads();                       // before next (o,d) restage
    }
  }

  const float* bv = bias + mb*64;
  float* Yb = Y + (size_t)mb*XPB + (size_t)b*(NN*NN*HH) + (size_t)mrow*4096;
#pragma unroll
  for (int ht = 0; ht < 4; ++ht)
#pragma unroll
    for (int v = 0; v < 8; ++v) {
      int e = w*16 + v + 8*hi;
      int h = ht*16 + lo;
      Yb[(size_t)e*64 + h] = acc[ht][v] + bv[h];
    }
}

// =====================================================================
// Kernel 4: FC (I=1) + ReLU + ensemble mean over branches.
// =====================================================================
__global__ __launch_bounds__(256) void fc_kernel(
    const float* __restrict__ Y2, const float* __restrict__ fcW,
    const float* __restrict__ fcb, float* __restrict__ out)
{
  int r = blockIdx.x*256 + threadIdx.x;
  if (r >= RTOT) return;
  float s = 0.0f;
#pragma unroll
  for (int mb = 0; mb < MM; ++mb) {
    const float* row = Y2 + (size_t)mb*XPB + (size_t)r*64;
    float a = fcb[mb];
#pragma unroll
    for (int h = 0; h < 64; ++h) a += row[h]*fcW[mb*64 + h];
    s += fmaxf(a, 0.0f);
  }
  out[r] = s * (1.0f/(float)MM);
}

// =====================================================================
extern "C" void kernel_launch(void* const* d_in, const int* in_sizes, int n_in,
                              void* d_out, int out_size, void* d_ws, size_t ws_size,
                              hipStream_t stream) {
  (void)in_sizes; (void)n_in; (void)out_size; (void)ws_size;
  const float* x_seq = (const float*)d_in[0];
  const float* G     = (const float*)d_in[1];
  const float* Wih   = (const float*)d_in[2];
  const float* Whh   = (const float*)d_in[3];
  const float* bihp  = (const float*)d_in[4];
  const float* bhhp  = (const float*)d_in[5];
  const float* W0    = (const float*)d_in[6];
  const float* b0    = (const float*)d_in[7];
  const float* W1    = (const float*)d_in[8];
  const float* b1    = (const float*)d_in[9];
  const float* fcW   = (const float*)d_in[10];
  const float* fcb   = (const float*)d_in[11];
  float* out = (float*)d_out;

  char* ws = (char*)d_ws;
  float*  Hbuf = (float*)ws;                                     // 2*XPB f32
  float*  Ybuf = (float*)(ws + (size_t)MM*XPB*sizeof(float));    // 2*XPB f32
  __bf16* T1g  = (__bf16*)(ws + (size_t)2*MM*XPB*sizeof(float)); // 2*24*64^3 bf16

  // LSTM -> Hbuf (per branch, layout (b, n, c, l))
  lstm_kernel<<<dim3(RTOT/(16*8), MM), 256, 0, stream>>>(
      x_seq, Wih, Whh, bihp, bhhp, Hbuf);

  // GCN layer 1: Hbuf -> T1g -> Ybuf
  gcn_t1_kernel<<<dim3(64, KK*BB, MM), 128, 0, stream>>>(Hbuf, G, T1g);
  gcn_out_kernel<<<dim3(NN, BB, MM), 128, 0, stream>>>(T1g, G, W0, b0, Ybuf);

  // GCN layer 2: Ybuf -> T1g -> Hbuf (Hbuf is dead, reuse as layer-2 output)
  gcn_t1_kernel<<<dim3(64, KK*BB, MM), 128, 0, stream>>>(Ybuf, G, T1g);
  gcn_out_kernel<<<dim3(NN, BB, MM), 128, 0, stream>>>(T1g, G, W1, b1, Hbuf);

  // FC + ReLU + ensemble mean -> d_out (32768 f32)
  fc_kernel<<<dim3(RTOT/256), 256, 0, stream>>>(Hbuf, fcW, fcb, out);
}
